// IndependentMLPs_36077725287197
// MI455X (gfx1250) — matrix operand order repaired and checked by reading, not compiled, not run on hardware
//
#include <hip/hip_runtime.h>
#include <hip/hip_bf16.h>

typedef __attribute__((ext_vector_type(8)))  _Float16 v8h;
typedef __attribute__((ext_vector_type(16))) _Float16 v16h;
typedef __attribute__((ext_vector_type(8)))  float    v8f;
typedef __attribute__((ext_vector_type(4)))  float    v4f;

#define B_DIM 1024
#define D_DIM 1024
#define O_DIM 1024
#define P_DIM 16

#define TILE_M 128
#define TILE_N 128
#define TILE_K 32
#define LDA 40   // padded LDS row stride in f16 (32 data + 8 pad) -> 80B, breaks 64B-stride conflicts
#define LDB 40

__global__ __launch_bounds__(256)
void IndependentMLPs_grouped_wmma_gelu(const float* __restrict__ x,
                                       const float* __restrict__ W,
                                       float* __restrict__ out)
{
    __shared__ __align__(16) _Float16 ldsA[TILE_M * LDA]; // A tile [m][k], k contiguous
    __shared__ __align__(16) _Float16 ldsB[TILE_N * LDB]; // B tile [n][k], k contiguous

    const int tid  = threadIdx.x;
    const int lane = tid & 31;
    const int wave = tid >> 5;

    const int p     = blockIdx.z;
    const int tileM = blockIdx.x * TILE_M;  // batch rows
    const int tileN = blockIdx.y * TILE_N;  // output cols (o)

    // 8 waves: 2 along M x 4 along N; each wave computes 64x32 = 4x2 WMMA tiles
    const int wm = wave & 1;
    const int wn = wave >> 1;

    // Loader mapping: each of 256 threads covers 16 elems of the A tile and 16 of the B tile
    const int lrow = tid >> 1;        // 0..127 (tile row for A, tile col-of-out for B)
    const int ldh  = (tid & 1) * 16;  // k sub-offset 0 or 16

    // x[b, d, p] at (b*D + d)*P + p : stride along d is P floats (64B) -> scalar loads, L2-resident
    const float* xrow = x + (size_t)(tileM + lrow) * D_DIM * P_DIM + p;
    // W[p, o, d] contiguous along d -> float4 loads
    const float* wrow = W + ((size_t)p * O_DIM + (tileN + lrow)) * D_DIM;

    float aReg[16];
    float bReg[16];

    // ---- prefetch K-tile 0 into registers ----
    {
        #pragma unroll
        for (int q = 0; q < 16; ++q)
            aReg[q] = xrow[(size_t)(ldh + q) * P_DIM];
        const v4f* wv = (const v4f*)(wrow + ldh);
        #pragma unroll
        for (int q4 = 0; q4 < 4; ++q4) {
            v4f v = wv[q4];
            bReg[q4 * 4 + 0] = v[0]; bReg[q4 * 4 + 1] = v[1];
            bReg[q4 * 4 + 2] = v[2]; bReg[q4 * 4 + 3] = v[3];
        }
    }

    v8f acc[4][2] = {};

    const int half = lane >> 4;   // K-half selector per ISA fragment layout
    const int l16  = lane & 15;

    const int NK = D_DIM / TILE_K; // 32
    for (int kt = 0; kt < NK; ++kt) {
        // ---- convert prefetched f32 -> f16 and stage into LDS ----
        {
            v8h a0, a1, b0, b1;
            #pragma unroll
            for (int q = 0; q < 8; ++q) {
                a0[q] = (_Float16)aReg[q];     a1[q] = (_Float16)aReg[8 + q];
                b0[q] = (_Float16)bReg[q];     b1[q] = (_Float16)bReg[8 + q];
            }
            _Float16* pa = &ldsA[lrow * LDA + ldh];
            *(v8h*)(pa)     = a0;
            *(v8h*)(pa + 8) = a1;
            _Float16* pb = &ldsB[lrow * LDB + ldh];
            *(v8h*)(pb)     = b0;
            *(v8h*)(pb + 8) = b1;
        }
        __syncthreads();

        // ---- prefetch next K-tile (overlaps with fragment loads + WMMA below) ----
        if (kt + 1 < NK) {
            const int d0 = (kt + 1) * TILE_K + ldh;
            #pragma unroll
            for (int q = 0; q < 16; ++q)
                aReg[q] = xrow[(size_t)(d0 + q) * P_DIM];
            const v4f* wv = (const v4f*)(wrow + d0);
            #pragma unroll
            for (int q4 = 0; q4 < 4; ++q4) {
                v4f v = wv[q4];
                bReg[q4 * 4 + 0] = v[0]; bReg[q4 * 4 + 1] = v[1];
                bReg[q4 * 4 + 2] = v[2]; bReg[q4 * 4 + 3] = v[3];
            }
        }

        // ---- preload ALL fragments into distinct registers, then an unbroken WMMA chain ----
        // B fragments: lane holds col N=l16, K chunks {h*8..+7, 16+h*8..+7}
        v16h bfrag[2];
        #pragma unroll
        for (int j = 0; j < 2; ++j) {
            const _Float16* pb = &ldsB[(wn * 32 + j * 16 + l16) * LDB + half * 8];
            v8h b0 = *(const v8h*)(pb);
            v8h b1 = *(const v8h*)(pb + 16);
            bfrag[j] = __builtin_shufflevector(b0, b1,
                0,1,2,3,4,5,6,7,8,9,10,11,12,13,14,15);
        }
        // A fragments: lane holds row M=l16, K chunks {h*8..+7, 16+h*8..+7}
        v16h afrag[4];
        #pragma unroll
        for (int i = 0; i < 4; ++i) {
            const _Float16* pa = &ldsA[(wm * 64 + i * 16 + l16) * LDA + half * 8];
            v8h a0 = *(const v8h*)(pa);
            v8h a1 = *(const v8h*)(pa + 16);
            afrag[i] = __builtin_shufflevector(a0, a1,
                0,1,2,3,4,5,6,7,8,9,10,11,12,13,14,15);
        }

        // 8 independent WMMAs back-to-back (no LDS waits interleaved)
        #pragma unroll
        for (int i = 0; i < 4; ++i) {
            #pragma unroll
            for (int j = 0; j < 2; ++j) {
                acc[i][j] = __builtin_amdgcn_wmma_f32_16x16x32_f16(
                    /*neg_a=*/false, afrag[i],
                    /*neg_b=*/false, bfrag[j],
                    /*c_mod=*/(short)0, acc[i][j],
                    /*reuse_a=*/false, /*reuse_b=*/false);
            }
        }
        __syncthreads();
    }

    // ---- epilogue: exact GELU in f32, scatter store (p is fastest output dim) ----
    // D layout: VGPR r -> M = half*8 + r, N = l16
    #pragma unroll
    for (int i = 0; i < 4; ++i) {
        const int bBase = tileM + wm * 64 + i * 16 + half * 8;
        #pragma unroll
        for (int j = 0; j < 2; ++j) {
            const int oIdx = tileN + wn * 32 + j * 16 + l16;
            #pragma unroll
            for (int r = 0; r < 8; ++r) {
                float v = acc[i][j][r];
                float g = 0.5f * v * (1.0f + erff(v * 0.7071067811865475f));
                out[((size_t)(bBase + r) * O_DIM + oIdx) * P_DIM + p] = g;
            }
        }
    }
}

extern "C" void kernel_launch(void* const* d_in, const int* in_sizes, int n_in,
                              void* d_out, int out_size, void* d_ws, size_t ws_size,
                              hipStream_t stream) {
    (void)in_sizes; (void)n_in; (void)out_size; (void)d_ws; (void)ws_size;
    const float* x = (const float*)d_in[0];   // [B, D, P] f32
    const float* W = (const float*)d_in[1];   // [P, O, D] f32
    float* out = (float*)d_out;               // [B, O, P] f32

    dim3 grid(B_DIM / TILE_M, O_DIM / TILE_N, P_DIM);
    dim3 block(256, 1, 1);
    IndependentMLPs_grouped_wmma_gelu<<<grid, block, 0, stream>>>(x, W, out);
}
